// DecoderRNN_69758858822113
// MI455X (gfx1250) — matrix operand order repaired and checked by reading, not compiled, not run on hardware
//
#include <hip/hip_runtime.h>
#include <math.h>

// Problem constants
constexpr int kB = 128, kP = 196, kEnc = 512, kEmb = 256, kDec = 512, kAtt = 256;
constexpr int kV = 2048, kMaxLen = 257, kT = kMaxLen - 1;
constexpr int kX = kEmb + kEnc + kDec;   // 1280: concat [emb | ctx | h]
constexpr int kG = 4 * kDec;             // 2048 gates

typedef __attribute__((ext_vector_type(16))) _Float16 v16h;
typedef __attribute__((ext_vector_type(8)))  _Float16 v8h;
typedef __attribute__((ext_vector_type(8)))  float    v8f;

// ---------------------------------------------------------------------------
// WMMA fragment loaders (gfx1250 wave32 layouts, CDNA5 ISA §7.12.2)
// ---------------------------------------------------------------------------

// A fragment: 16x32 f16, row-major f16 source.
// lane<16: M=lane, K = {0..7} in a[0..7], {16..23} in a[8..15]
// lane>=16: M=lane-16, K = {8..15} in a[0..7], {24..31} in a[8..15]
__device__ inline v16h load_a_frag_f16(const _Float16* __restrict__ Arow, int lda,
                                       int k0, int lane) {
  int m  = lane & 15;
  int kb = (lane >> 4) << 3;           // 0 or 8
  const _Float16* p = Arow + (size_t)m * lda + k0 + kb;
  v8h x = *(const v8h*)(p);            // K = kb+0 .. kb+7
  v8h y = *(const v8h*)(p + 16);       // K = kb+16 .. kb+23
  v16h a;
#pragma unroll
  for (int j = 0; j < 8; ++j) { a[j] = x[j]; a[8 + j] = y[j]; }
  return a;
}

// B fragment: 32x16 f16 (B element (k,n) = W[n,k], W row-major N x K).
// lane<16: N=lane, K={0..7,16..23}; lane>=16: N=lane-16, K={8..15,24..31}.
__device__ inline v16h load_b_frag_f16(const _Float16* __restrict__ W, int K,
                                       int col0, int k0, int lane) {
  int n  = col0 + (lane & 15);
  int kb = (lane >> 4) << 3;           // 0 or 8
  const _Float16* p = W + (size_t)n * K + k0 + kb;
  v8h b0 = *(const v8h*)(p);
  v8h b1 = *(const v8h*)(p + 16);
  v16h b;
#pragma unroll
  for (int j = 0; j < 8; ++j) { b[j] = b0[j]; b[8 + j] = b1[j]; }
  return b;
}

__device__ inline v8f wmma_step(v16h a, v16h b, v8f c) {
  return __builtin_amdgcn_wmma_f32_16x16x32_f16(
      /*neg_a=*/false, a, /*neg_b=*/false, b,
      /*c_mod=*/(short)0, c, /*reuse_a=*/false, /*reuse_b=*/false);
}

// ---------------------------------------------------------------------------
// WMMA GEMM, 2x2 tiles per wave: out[m,n] = A16[m,:] . W16[n,:] + bias[n].
// Block = 256 threads = 8 waves; wave tile = 32 rows x 32 cols;
// block tile = 32 rows x 256 cols.  Grid: x = N/256, y = M/32.
// Optional f32 output (with optional per-row mask t < cl[row]-1, zeros
// masked rows) and optional f16 output (unmasked).
// ---------------------------------------------------------------------------
__global__ __launch_bounds__(256)
void wmma_gemm_2x2(const _Float16* __restrict__ A, int lda,
                   const _Float16* __restrict__ W, int K, int N,
                   const float* __restrict__ bias,
                   float* __restrict__ out, _Float16* __restrict__ out16,
                   long long row_stride, long long col_off,
                   const int* __restrict__ cl, int t) {
  int lane = threadIdx.x & 31;
  int wave = threadIdx.x >> 5;
  int row0 = blockIdx.y << 5;                  // 32 rows per block
  int col0 = (blockIdx.x << 8) + (wave << 5);  // 32 cols per wave
  if (col0 >= N) return;

  const _Float16* Arow0 = A + (size_t)row0 * lda;
  const _Float16* Arow1 = Arow0 + (size_t)16 * lda;

  v8f acc[2][2] = {};
  for (int k0 = 0; k0 < K; k0 += 32) {
    v16h a0 = load_a_frag_f16(Arow0, lda, k0, lane);
    v16h a1 = load_a_frag_f16(Arow1, lda, k0, lane);
    v16h b0 = load_b_frag_f16(W, K, col0, k0, lane);
    v16h b1 = load_b_frag_f16(W, K, col0 + 16, k0, lane);
    acc[0][0] = wmma_step(a0, b0, acc[0][0]);
    acc[0][1] = wmma_step(a0, b1, acc[0][1]);
    acc[1][0] = wmma_step(a1, b0, acc[1][0]);
    acc[1][1] = wmma_step(a1, b1, acc[1][1]);
  }

#pragma unroll
  for (int ct = 0; ct < 2; ++ct) {
    int col  = col0 + ct * 16 + (lane & 15);
    float bv = bias ? bias[col] : 0.0f;
#pragma unroll
    for (int rt = 0; rt < 2; ++rt) {
      int rbase = row0 + rt * 16 + ((lane >> 4) << 3);
#pragma unroll
      for (int r = 0; r < 8; ++r) {
        int row = rbase + r;
        float v = acc[rt][ct][r] + bv;
        if (out) {
          float vm = (cl && t >= cl[row] - 1) ? 0.0f : v;
          out[(long long)row * row_stride + col_off + col] = vm;
        }
        if (out16) {
          out16[(long long)row * row_stride + col_off + col] = (_Float16)v;
        }
      }
    }
  }
}

// ---------------------------------------------------------------------------
// Setup helpers
// ---------------------------------------------------------------------------
__global__ void f32_to_f16_kernel(const float* __restrict__ src,
                                  _Float16* __restrict__ dst, int n) {
  int i = blockIdx.x * blockDim.x + threadIdx.x;
  if (i < n) dst[i] = (_Float16)src[i];
}

// Wcat[n, 0:768] = W_ih[n, :], Wcat[n, 768:1280] = W_hh[n, :]
__global__ void build_wcat_kernel(const float* __restrict__ W_ih,
                                  const float* __restrict__ W_hh,
                                  _Float16* __restrict__ Wcat) {
  int i = blockIdx.x * blockDim.x + threadIdx.x;
  if (i >= kG * kX) return;
  int n = i / kX, k = i % kX;
  float v = (k < kEmb + kEnc) ? W_ih[(size_t)n * (kEmb + kEnc) + k]
                              : W_hh[(size_t)n * kDec + (k - kEmb - kEnc)];
  Wcat[i] = (_Float16)v;
}

__global__ void bias2_kernel(const float* __restrict__ a,
                             const float* __restrict__ b,
                             float* __restrict__ o, int n) {
  int i = blockIdx.x * blockDim.x + threadIdx.x;
  if (i < n) o[i] = a[i] + b[i];
}

// mean over P of encoder_out: (B, P, ENC) -> f16 (B, ENC)
__global__ void mean_enc_kernel(const float* __restrict__ enc,
                                _Float16* __restrict__ mean16) {
  int i = blockIdx.x * blockDim.x + threadIdx.x;
  if (i >= kB * kEnc) return;
  int b = i / kEnc, e = i % kEnc;
  const float* p = enc + (size_t)b * kP * kEnc + e;
  float s = 0.f;
  for (int pp = 0; pp < kP; ++pp) s += p[(size_t)pp * kEnc];
  mean16[i] = (_Float16)(s * (1.0f / (float)kP));
}

// ---------------------------------------------------------------------------
// Per-step kernels
// ---------------------------------------------------------------------------

// e[b,p] = relu(att1[b,p,:] + att2[b,:]) . Wfull + bfull
__global__ void att_e_kernel(const float* __restrict__ att1,
                             const float* __restrict__ att2,
                             const float* __restrict__ wfull,
                             const float* __restrict__ bfull,
                             float* __restrict__ e) {
  int idx = blockIdx.x * blockDim.x + threadIdx.x;
  if (idx >= kB * kP) return;
  int b = idx / kP;
  const float4* a1 = (const float4*)(att1 + (size_t)idx * kAtt);
  const float4* a2 = (const float4*)(att2 + (size_t)b * kAtt);
  const float4* wf = (const float4*)(wfull);
  float s = 0.f;
#pragma unroll 4
  for (int i = 0; i < kAtt / 4; ++i) {
    float4 x = a1[i], y = a2[i], w = wf[i];
    s = fmaf(fmaxf(x.x + y.x, 0.f), w.x, s);
    s = fmaf(fmaxf(x.y + y.y, 0.f), w.y, s);
    s = fmaf(fmaxf(x.z + y.z, 0.f), w.z, s);
    s = fmaf(fmaxf(x.w + y.w, 0.f), w.w, s);
  }
  e[idx] = s + bfull[0];
}

// softmax over P per batch; writes alpha to ws and masked alpha to output
__global__ __launch_bounds__(256)
void softmax_alpha_kernel(const float* __restrict__ e,
                          float* __restrict__ alpha,
                          float* __restrict__ alpha_out,
                          const int* __restrict__ cl, int t) {
  __shared__ float red[256];
  int b = blockIdx.x, tid = threadIdx.x;
  float v = (tid < kP) ? e[b * kP + tid] : -3.4e38f;
  red[tid] = v;
  __syncthreads();
  for (int s = 128; s > 0; s >>= 1) {
    if (tid < s) red[tid] = fmaxf(red[tid], red[tid + s]);
    __syncthreads();
  }
  float mx = red[0];
  __syncthreads();
  float ex = (tid < kP) ? expf(v - mx) : 0.f;
  red[tid] = ex;
  __syncthreads();
  for (int s = 128; s > 0; s >>= 1) {
    if (tid < s) red[tid] += red[tid + s];
    __syncthreads();
  }
  float inv = 1.0f / red[0];
  if (tid < kP) {
    float a = ex * inv;
    alpha[b * kP + tid] = a;
    bool mk = t < (cl[b] - 1);
    alpha_out[((long long)b * kT + t) * kP + tid] = mk ? a : 0.0f;
  }
}

// ctx[b,e] = sum_p alpha[b,p] * enc[b,p,e]  -> x16[b, kEmb + e]
__global__ void ctx_kernel(const float* __restrict__ alpha,
                           const float* __restrict__ enc,
                           _Float16* __restrict__ x16) {
  int idx = blockIdx.x * blockDim.x + threadIdx.x;
  if (idx >= kB * kEnc) return;
  int b = idx / kEnc, e = idx % kEnc;
  const float* al = alpha + b * kP;
  const float* ep = enc + (size_t)b * kP * kEnc + e;
  float s = 0.f;
  for (int p = 0; p < kP; ++p) s = fmaf(al[p], ep[(size_t)p * kEnc], s);
  x16[(size_t)b * kX + kEmb + e] = (_Float16)s;
}

// x16[b, 0:256] = embedding[cap[b,t]]; x16[b, 768:1280] = h16[b,:]
__global__ void prep_x_kernel(const float* __restrict__ embedding,
                              const int* __restrict__ caps,
                              const _Float16* __restrict__ h16,
                              _Float16* __restrict__ x16, int t) {
  int idx = blockIdx.x * blockDim.x + threadIdx.x;
  if (idx >= kB * (kEmb + kDec)) return;
  int b = idx / (kEmb + kDec), j = idx % (kEmb + kDec);
  _Float16* xr = x16 + (size_t)b * kX;
  if (j < kEmb) {
    int tok = caps[b * kMaxLen + t];
    xr[j] = (_Float16)embedding[(size_t)tok * kEmb + j];
  } else {
    xr[kEmb + kEnc + (j - kEmb)] = h16[b * kDec + (j - kEmb)];
  }
}

__device__ inline float sigmoidf(float x) { return 1.0f / (1.0f + expf(-x)); }

// masked LSTM state update: gates order [i | f | g | o]; h kept in f16
__global__ void lstm_update_kernel(const float* __restrict__ gates,
                                   _Float16* __restrict__ h16,
                                   float* __restrict__ c,
                                   const int* __restrict__ cl, int t) {
  int idx = blockIdx.x * blockDim.x + threadIdx.x;
  if (idx >= kB * kDec) return;
  int b = idx / kDec, d = idx % kDec;
  if (t >= cl[b] - 1) return;   // mask: only update where t < decode_len
  const float* g = gates + (size_t)b * kG;
  float ig = g[d], fg = g[kDec + d], gg = g[2 * kDec + d], og = g[3 * kDec + d];
  float cn = sigmoidf(fg) * c[idx] + sigmoidf(ig) * tanhf(gg);
  float hn = sigmoidf(og) * tanhf(cn);
  c[idx]  = cn;
  h16[idx] = (_Float16)hn;
}

// ---------------------------------------------------------------------------
// Host launcher
// ---------------------------------------------------------------------------
extern "C" void kernel_launch(void* const* d_in, const int* in_sizes, int n_in,
                              void* d_out, int out_size, void* d_ws, size_t ws_size,
                              hipStream_t stream) {
  (void)in_sizes; (void)n_in; (void)out_size; (void)ws_size;

  const float* enc       = (const float*)d_in[0];
  const int*   caps      = (const int*)d_in[1];
  const int*   cl        = (const int*)d_in[2];
  const float* embedding = (const float*)d_in[3];
  const float* W_enc_att = (const float*)d_in[4];
  const float* b_enc_att = (const float*)d_in[5];
  const float* W_dec_att = (const float*)d_in[6];
  const float* b_dec_att = (const float*)d_in[7];
  const float* W_full    = (const float*)d_in[8];
  const float* b_full    = (const float*)d_in[9];
  const float* W_ih      = (const float*)d_in[10];
  const float* b_ih      = (const float*)d_in[11];
  const float* W_hh      = (const float*)d_in[12];
  const float* b_hh      = (const float*)d_in[13];
  const float* W_init_h  = (const float*)d_in[14];
  const float* b_init_h  = (const float*)d_in[15];
  const float* W_init_c  = (const float*)d_in[16];
  const float* b_init_c  = (const float*)d_in[17];
  const float* W_fc      = (const float*)d_in[18];
  const float* b_fc      = (const float*)d_in[19];

  float* preds_out  = (float*)d_out;
  float* alphas_out = preds_out + (size_t)kB * kT * kV;

  // workspace layout
  char* base = (char*)d_ws;
  size_t off = 0;
  auto alloc = [&](size_t bytes) -> void* {
    void* p = base + off;
    off = (off + bytes + 255) & ~(size_t)255;
    return p;
  };
  _Float16* Wenc16 = (_Float16*)alloc(sizeof(_Float16) * kAtt * kEnc);
  _Float16* Wdec16 = (_Float16*)alloc(sizeof(_Float16) * kAtt * kDec);
  _Float16* Winh16 = (_Float16*)alloc(sizeof(_Float16) * kDec * kEnc);
  _Float16* Winc16 = (_Float16*)alloc(sizeof(_Float16) * kDec * kEnc);
  _Float16* Wcat16 = (_Float16*)alloc(sizeof(_Float16) * kG * kX);
  _Float16* Wfc16  = (_Float16*)alloc(sizeof(_Float16) * kV * kDec);
  _Float16* enc16  = (_Float16*)alloc(sizeof(_Float16) * (size_t)kB * kP * kEnc);
  _Float16* mean16 = (_Float16*)alloc(sizeof(_Float16) * kB * kEnc);
  _Float16* h16    = (_Float16*)alloc(sizeof(_Float16) * kB * kDec);
  _Float16* x16    = (_Float16*)alloc(sizeof(_Float16) * kB * kX);
  float* bcat  = (float*)alloc(sizeof(float) * kG);
  float* cbuf  = (float*)alloc(sizeof(float) * kB * kDec);
  float* att1  = (float*)alloc(sizeof(float) * (size_t)kB * kP * kAtt);
  float* att2  = (float*)alloc(sizeof(float) * kB * kAtt);
  float* ebuf  = (float*)alloc(sizeof(float) * kB * kP);
  float* alphab= (float*)alloc(sizeof(float) * kB * kP);
  float* gates = (float*)alloc(sizeof(float) * kB * kG);

  const int TPB = 256;
  auto cdiv = [](int a, int b) { return (a + b - 1) / b; };

  // ---- setup: weight / activation conversion ----
  f32_to_f16_kernel<<<cdiv(kAtt * kEnc, TPB), TPB, 0, stream>>>(W_enc_att, Wenc16, kAtt * kEnc);
  f32_to_f16_kernel<<<cdiv(kAtt * kDec, TPB), TPB, 0, stream>>>(W_dec_att, Wdec16, kAtt * kDec);
  f32_to_f16_kernel<<<cdiv(kDec * kEnc, TPB), TPB, 0, stream>>>(W_init_h, Winh16, kDec * kEnc);
  f32_to_f16_kernel<<<cdiv(kDec * kEnc, TPB), TPB, 0, stream>>>(W_init_c, Winc16, kDec * kEnc);
  f32_to_f16_kernel<<<cdiv(kV * kDec, TPB), TPB, 0, stream>>>(W_fc, Wfc16, kV * kDec);
  f32_to_f16_kernel<<<cdiv(kB * kP * kEnc, TPB), TPB, 0, stream>>>(enc, enc16, kB * kP * kEnc);
  build_wcat_kernel<<<cdiv(kG * kX, TPB), TPB, 0, stream>>>(W_ih, W_hh, Wcat16);
  bias2_kernel<<<cdiv(kG, TPB), TPB, 0, stream>>>(b_ih, b_hh, bcat, kG);

  // ---- setup: h0 (f16), c0 (f32), att1 ----
  mean_enc_kernel<<<cdiv(kB * kEnc, TPB), TPB, 0, stream>>>(enc, mean16);
  wmma_gemm_2x2<<<dim3(cdiv(kDec, 256), kB / 32), 256, 0, stream>>>(
      mean16, kEnc, Winh16, kEnc, kDec, b_init_h, nullptr, h16, kDec, 0, nullptr, 0);
  wmma_gemm_2x2<<<dim3(cdiv(kDec, 256), kB / 32), 256, 0, stream>>>(
      mean16, kEnc, Winc16, kEnc, kDec, b_init_c, cbuf, nullptr, kDec, 0, nullptr, 0);
  wmma_gemm_2x2<<<dim3(cdiv(kAtt, 256), (kB * kP) / 32), 256, 0, stream>>>(
      enc16, kEnc, Wenc16, kEnc, kAtt, b_enc_att, att1, nullptr, kAtt, 0, nullptr, 0);

  // ---- recurrent loop ----
  for (int t = 0; t < kT; ++t) {
    wmma_gemm_2x2<<<dim3(cdiv(kAtt, 256), kB / 32), 256, 0, stream>>>(
        h16, kDec, Wdec16, kDec, kAtt, b_dec_att, att2, nullptr, kAtt, 0, nullptr, 0);
    att_e_kernel<<<cdiv(kB * kP, TPB), TPB, 0, stream>>>(att1, att2, W_full, b_full, ebuf);
    softmax_alpha_kernel<<<kB, 256, 0, stream>>>(ebuf, alphab, alphas_out, cl, t);
    ctx_kernel<<<cdiv(kB * kEnc, TPB), TPB, 0, stream>>>(alphab, enc, x16);
    prep_x_kernel<<<cdiv(kB * (kEmb + kDec), TPB), TPB, 0, stream>>>(embedding, caps, h16, x16, t);
    wmma_gemm_2x2<<<dim3(cdiv(kG, 256), kB / 32), 256, 0, stream>>>(
        x16, kX, Wcat16, kX, kG, bcat, gates, nullptr, kG, 0, nullptr, 0);
    lstm_update_kernel<<<cdiv(kB * kDec, TPB), TPB, 0, stream>>>(gates, h16, cbuf, cl, t);
    wmma_gemm_2x2<<<dim3(cdiv(kV, 256), kB / 32), 256, 0, stream>>>(
        h16, kDec, Wfc16, kDec, kV, b_fc, preds_out, nullptr,
        (long long)kT * kV, (long long)t * kV, cl, t);
  }
}
